// Word2VecCBOWModel_55490977464738
// MI455X (gfx1250) — matrix operand order repaired and checked by reading, not compiled, not run on hardware
//
#include <hip/hip_runtime.h>

#define VOCAB 50000
#define DIM   300
#define KP    320      // K padded to 10 x 32 for bf16 WMMA
#define BATCH 1024
#define CTX   10
#define LDA   328      // LDS row pitch (halves): 328*2B = 656B = 164 dwords,
                       // 164 % 64 = 36 -> 16 rows hit 16 distinct banks

typedef __attribute__((ext_vector_type(16))) __bf16 v16bf;
typedef __attribute__((ext_vector_type(8)))  __bf16 v8bf;
typedef __attribute__((ext_vector_type(8)))  float  v8f;

__device__ __forceinline__ unsigned short f32_to_bf16(float f) {
    union { float f; unsigned int u; } v; v.f = f;
    unsigned int b = v.u;
    unsigned int r = b + 0x7FFFu + ((b >> 16) & 1u);   // round-to-nearest-even
    if ((b & 0x7F800000u) == 0x7F800000u) r = b;       // inf/nan: truncate
    return (unsigned short)(r >> 16);
}

// ---------------------------------------------------------------------------
// Kernel 1: masked-mean pooling of embeddings -> bf16 pooled [BATCH, KP]
// ---------------------------------------------------------------------------
__global__ __launch_bounds__(KP)
void pool_kernel(const int* __restrict__ ctx, const float* __restrict__ Wproj,
                 const int* __restrict__ pad_p, unsigned short* __restrict__ Pb) {
    const int b = blockIdx.x;
    const int d = threadIdx.x;                 // 0..319
    const int pad = pad_p[0];
    float s = 0.f;
    if (d < DIM) {
        #pragma unroll
        for (int c = 0; c < CTX; ++c) {
            int id = ctx[b * CTX + c];
            if (id != pad) s += Wproj[(size_t)d * VOCAB + id];
        }
        s *= (1.0f / CTX);
    }
    Pb[(size_t)b * KP + d] = f32_to_bf16(d < DIM ? s : 0.f);
}

// ---------------------------------------------------------------------------
// Kernel 2: W_cls f32 [VOCAB, DIM] -> bf16 [VOCAB, KP] (zero-padded K tail)
// ---------------------------------------------------------------------------
__global__ __launch_bounds__(256)
void cvt_wcls_kernel(const float* __restrict__ Wcls, unsigned short* __restrict__ Wb) {
    const size_t total = (size_t)VOCAB * KP;
    for (size_t i = (size_t)blockIdx.x * blockDim.x + threadIdx.x; i < total;
         i += (size_t)gridDim.x * blockDim.x) {
        int k = (int)(i % KP);
        size_t v = i / KP;
        Wb[i] = f32_to_bf16(k < DIM ? Wcls[v * DIM + k] : 0.f);
    }
}

// ---------------------------------------------------------------------------
// Kernel 3: logits = pooled @ W_cls^T + b_cls via v_wmma_f32_16x16x32_bf16.
// Block: 16 waves (512 thr). The 80-row W_cls strip for this block's vocab
// super-tile is staged to LDS once (51 KB) and shared by all 16 waves
// (16 m-tiles = 256 batch rows). Per k-step: all 5 A-fragments are loaded
// into distinct registers first (one ds clause, one wait), then the 5 WMMAs
// issue back-to-back on independent accumulator chains, so the next k-step's
// loads overlap the current WMMAs.
// ---------------------------------------------------------------------------
__global__ __launch_bounds__(512)
void gemm_wmma_kernel(const unsigned short* __restrict__ Pb,
                      const unsigned short* __restrict__ Wb,
                      const float* __restrict__ bcls,
                      float* __restrict__ out) {
    __shared__ __align__(16) unsigned short lA[80 * LDA];

    const int tid  = threadIdx.x;
    const int lane = tid & 31;
    const int wave = tid >> 5;
    const int n0 = blockIdx.x * 80;                 // vocab super-tile (5 x 16)
    const int m0 = (blockIdx.y * 16 + wave) * 16;   // batch tile per wave

    // Stage A strip: 80 rows x 320 halves (b128 chunks), padded LDS pitch.
    for (int i = tid; i < 80 * 40; i += 512) {
        const int r = i / 40, c = (i % 40) * 8;
        *(v8bf*)(lA + r * LDA + c) = *(const v8bf*)(Wb + (size_t)(n0 + r) * KP + c);
    }
    __syncthreads();

    v8f acc[5] = {};

    const int l15 = lane & 15;
    const int hiA = (lane >> 4) * 16;   // A-frag K sub-offset per lane half
    const int hiB = (lane >> 4) * 16;   // B-frag K sub-offset per lane half

    const unsigned short* prow = Pb + (size_t)(m0 + l15) * KP + hiB;
    const unsigned short* arow = lA + l15 * LDA + hiA;

    for (int k0 = 0; k0 < KP; k0 += 32) {
        // B fragment: 16 contiguous bf16 (32B) per lane from global (L2-hit)
        v16bf bfrag = *(const v16bf*)(prow + k0);
        // All 5 A fragments up-front: 10 ds_load_b128 in one clause
        v16bf afr[5];
        #pragma unroll
        for (int t = 0; t < 5; ++t) {
            const unsigned short* ar = arow + t * (16 * LDA) + k0;
            ((v8bf*)&afr[t])[0] = *(const v8bf*)(ar);        // K 0..7  (lo half)
            ((v8bf*)&afr[t])[1] = *(const v8bf*)(ar + 8);    // K 8..15
        }
        // 5 back-to-back WMMAs on independent accumulators
        #pragma unroll
        for (int t = 0; t < 5; ++t) {
            acc[t] = __builtin_amdgcn_wmma_f32_16x16x32_bf16(
                false, afr[t], false, bfrag, (short)0, acc[t], false, false);
        }
    }

    // Store: lane holds 8 consecutive logits per tile; lanes l / l+16 cover
    // cols 0..7 / 8..15 of the same row -> 320B contiguous per row per wave.
    const int m = m0 + l15;
    const int coff = (lane >> 4) * 8;
    #pragma unroll
    for (int t = 0; t < 5; ++t) {
        const int n = n0 + t * 16 + coff;
        v8f bias = *(const v8f*)(bcls + n);
        *(v8f*)(out + (size_t)m * VOCAB + n) = acc[t] + bias;
    }
}

// ---------------------------------------------------------------------------
// Kernel 4: in-place row-wise log_softmax (online max/sumexp; 2 reads +
// 1 write over the 205 MB output).
// ---------------------------------------------------------------------------
__global__ __launch_bounds__(256)
void logsoftmax_kernel(float* __restrict__ out) {
    __shared__ float smax[256];
    __shared__ float ssum[256];
    float* p = out + (size_t)blockIdx.x * VOCAB;
    const int tid = threadIdx.x;

    float m = -3.4e38f, s = 0.f;
    for (int n = tid; n < VOCAB; n += 256) {
        float x = p[n];
        if (x > m) { s = s * __expf(m - x) + 1.f; m = x; }
        else       { s += __expf(x - m); }
    }
    smax[tid] = m; ssum[tid] = s;
    __syncthreads();
    for (int off = 128; off > 0; off >>= 1) {
        if (tid < off) {
            float m1 = smax[tid], s1 = ssum[tid];
            float m2 = smax[tid + off], s2 = ssum[tid + off];
            float M = fmaxf(m1, m2);
            smax[tid] = M;
            ssum[tid] = s1 * __expf(m1 - M) + s2 * __expf(m2 - M);
        }
        __syncthreads();
    }
    const float lse = smax[0] + __logf(ssum[0]);
    for (int n = tid; n < VOCAB; n += 256) p[n] -= lse;
}

extern "C" void kernel_launch(void* const* d_in, const int* in_sizes, int n_in,
                              void* d_out, int out_size, void* d_ws, size_t ws_size,
                              hipStream_t stream) {
    const int*   ctx   = (const int*)d_in[0];
    const float* Wproj = (const float*)d_in[1];
    const float* Wcls  = (const float*)d_in[2];
    const float* bcls  = (const float*)d_in[3];
    const int*   pad   = (const int*)d_in[4];
    float* out = (float*)d_out;

    // workspace: bf16 pooled [1024,320] (640 KB) then bf16 W_cls [50000,320] (32 MB)
    unsigned short* Pb = (unsigned short*)d_ws;
    unsigned short* Wb = Pb + (size_t)BATCH * KP;

    cvt_wcls_kernel<<<2048, 256, 0, stream>>>(Wcls, Wb);
    pool_kernel<<<BATCH, KP, 0, stream>>>(ctx, Wproj, pad, Pb);
    dim3 grid(VOCAB / 80, BATCH / (16 * 16));   // (625, 4), 16 waves/block
    gemm_wmma_kernel<<<grid, 512, 0, stream>>>(Pb, Wb, bcls, out);
    logsoftmax_kernel<<<BATCH, 256, 0, stream>>>(out);
}
